// HyperRandomGaussianMixture_44074954392170
// MI455X (gfx1250) — matrix operand order repaired and checked by reading, not compiled, not run on hardware
//
#include <hip/hip_runtime.h>
#include <hip/hip_bf16.h>
#include <math.h>

// ---------------------------------------------------------------------------
// HyperRandomGaussianMixture on MI455X (gfx1250, wave32, WMMA)
//
// Separable smoothing = banded 96x96 matmul per axis. Folded (symmetric-pad)
// smoothing matrix S has row support [i-7, i+7], so a 16-row output tile
// needs a K-window of width 30 <= 32: ONE v_wmma_f32_16x16x32_f16 per
// 16x16 tile with C = 0.
//
// Pipeline (2 volume kernels):
//   fused12: per (vol, a0) workgroup (288 thr = 9 waves), one 96x96 (a1,a2)
//            plane in LDS:
//       global fp32 noise -> affine(means,scales) -> LDS f16 P0[a1][a2]
//       smooth a2 (WMMA, ds_load_b128 B-frags) -> LDS P1[a2][a1]
//       smooth a1 (WMMA)                       -> global t[a1][a2][a0] (f16)
//       9 waves x 4 tiles = 36 tiles per stage, fully uniform loops
//       (no EXEC-mask divergence loop; WMMA needs EXEC == all-1s).
//   pass3:   smooth a0 (contiguous in t) + fused einsum over K -> fp32 out.
//
// HBM traffic ~382 MB (t round-trip mostly L2-resident: 70.8MB < 192MB L2)
// => ~13-16 us floor at 23.3 TB/s; ~415k f16 WMMAs are far below the matrix
// ceiling, so the whole pipeline is bandwidth-bound.
//
// d_ws layout:  [0, 48KB)        folded S tiles, f16, (b,c,I=0..5) x 16x32
//               [64KB, +70.8MB)  t  (B*C*K*96^3 f16, layout [a1][a2][a0])
// ---------------------------------------------------------------------------

typedef _Float16 v16h __attribute__((ext_vector_type(16)));
typedef _Float16 v8h  __attribute__((ext_vector_type(8)));
typedef _Float16 v4h  __attribute__((ext_vector_type(4)));
typedef float    v8f  __attribute__((ext_vector_type(8)));
typedef float    v4f  __attribute__((ext_vector_type(4)));

#define NB 4
#define NC 2
#define NK 5
#define NE 96                 // edge
#define NV 884736             // 96^3
#define S_YZ 9216             // 96^2
#define NTILE 6               // 96 / 16
#define ATILE_HALFS 512       // 16x32 f16 per tile

__device__ __forceinline__ int win0(int I) {
    int w = 16 * I - 8;
    return w < 0 ? 0 : (w > 64 ? 64 : w);
}

__device__ __forceinline__ int reflect96(int p) {
    p = (p < 0) ? (-p - 1) : p;
    p = (p > 95) ? (191 - p) : p;
    return p;
}

// ---------------------------------------------------------------------------
// Kernel 0: build folded smoothing-matrix tiles per (b,c), f16.
//   A[I][m][kk] = S[16I+m][win0(I)+kk],
//   S[i][j] = sum_t kern[t] * (reflect(i + t - 7) == j)
// ---------------------------------------------------------------------------
__global__ void build_S_kernel(const float* __restrict__ fwhm_eps,
                               _Float16* __restrict__ wsA) {
    const int bc = blockIdx.x;                     // 0..7
    __shared__ float kern[15];
    if (threadIdx.x == 0) {
        // fwhm ~ LogNormal(mean=3, sd=5)
        const float s2 = logf(1.0f + (5.0f / 3.0f) * (5.0f / 3.0f));
        const float mu = logf(3.0f) - 0.5f * s2;
        const float fw = expf(mu + sqrtf(s2) * fwhm_eps[bc]);
        const float sigma = fw * 0.42466090f;      // 1/sqrt(8 ln 2)
        float tmp[15], sum = 0.0f;
        for (int t = 0; t < 15; ++t) {
            float d = (float)(t - 7) / sigma;
            tmp[t] = expf(-0.5f * d * d);
            sum += tmp[t];
        }
        for (int t = 0; t < 15; ++t) kern[t] = tmp[t] / sum;
    }
    __syncthreads();
    for (int idx = threadIdx.x; idx < NTILE * 16 * 32; idx += blockDim.x) {
        const int I  = idx >> 9;
        const int rm = idx & 511;
        const int m  = rm >> 5;
        const int kk = rm & 31;
        const int i  = 16 * I + m;
        const int j  = win0(I) + kk;
        float v = 0.0f;
        #pragma unroll
        for (int t = 0; t < 15; ++t)
            if (reflect96(i + t - 7) == j) v += kern[t];
        wsA[bc * (NTILE * ATILE_HALFS) + idx] = (_Float16)v;
    }
}

// A fragment (16x32 f16), row-major tile in ws.
// Lane L: M = L%16, half = L/16. elems 0..15 -> K = 16*half + e.
__device__ __forceinline__ v16h load_afrag(const _Float16* __restrict__ tile,
                                           int lane) {
    const int N = lane & 15, half = lane >> 4;
    const _Float16* row = tile + N * 32;
    v8h lo = *(const v8h*)(row + 16 * half);
    v8h hi = *(const v8h*)(row + 16 * half + 8);
    return __builtin_shufflevector(lo, hi, 0, 1, 2, 3, 4, 5, 6, 7,
                                           8, 9, 10, 11, 12, 13, 14, 15);
}

__device__ __forceinline__ v16h combine16(v8h lo, v8h hi) {
    return __builtin_shufflevector(lo, hi, 0, 1, 2, 3, 4, 5, 6, 7,
                                           8, 9, 10, 11, 12, 13, 14, 15);
}

// ---------------------------------------------------------------------------
// Fused pass 1+2: per (vol, a0) plane. 3840 WGs x 288 threads (9 waves).
//   step0: noise plane (a1,a2) fp32 -> affine -> P0[a1][a2] f16 (LDS)
//   stepA: smooth a2: B-frag contiguous from P0, D -> P1[a2][a1] (LDS)
//   stepB: smooth a1: B-frag contiguous from P1, D -> t[a1][a2][a0] (global)
// 36 tiles per stage, 4 per wave: uniform trip count, EXEC stays all-1s.
// ---------------------------------------------------------------------------
__global__ __launch_bounds__(288) void fused12_kernel(
        const float* __restrict__ noise,
        const float* __restrict__ means_eps,
        const float* __restrict__ scales_sample,
        const _Float16* __restrict__ wsA,
        _Float16* __restrict__ t) {
    __shared__ _Float16 P0[S_YZ];   // [a1][a2], 18KB
    __shared__ _Float16 P1[S_YZ];   // [a2][a1], 18KB

    const int tid  = threadIdx.x;
    const int lane = tid & 31;
    const int wave = tid >> 5;                     // 0..8
    const int a0   = blockIdx.x % NE;
    const int vol  = blockIdx.x / NE;              // (b*C+c)*K + k
    const int bc   = vol / NK;

    const float mmu = 10.0f * means_eps[vol];
    const float ssc = fmaxf(scales_sample[vol], 0.0f);

    // ---- step0: stage affine-transformed plane into LDS (f16) ----
    const float* np = noise + (size_t)vol * NV + (size_t)a0 * S_YZ;
    #pragma unroll
    for (int it = 0; it < 8; ++it) {               // 2304 float4 / 288 thr
        const int idx4 = tid + 288 * it;
        const v4f f = *(const v4f*)(np + idx4 * 4);
        v4h h;
        #pragma unroll
        for (int i = 0; i < 4; ++i) h[i] = (_Float16)fmaf(ssc, f[i], mmu);
        *(v4h*)(P0 + idx4 * 4) = h;
    }
    __syncthreads();

    const int N = lane & 15, half = lane >> 4;
    const _Float16* Abc = wsA + bc * (NTILE * ATILE_HALFS);

    // ---- stepA: smooth a2.  36 tiles: (I = a2-tile, a1g) ----
    #pragma unroll
    for (int it = 0; it < 4; ++it) {
        const int tt = wave + 9 * it;              // 0..35, uniform trip count
        const int I = tt / 6, a1g = tt % 6;
        const int w0 = win0(I);
        const v16h afrag = load_afrag(Abc + I * ATILE_HALFS, lane);
        const _Float16* p = P0 + (a1g * 16 + N) * NE + w0 + 16 * half;
        const v16h bfrag = combine16(*(const v8h*)p, *(const v8h*)(p + 8));
        v8f cz = {};
        v8f d = __builtin_amdgcn_wmma_f32_16x16x32_f16(
                    false, afrag, false, bfrag, (short)0, cz, false, false);
        _Float16* dst = P1 + (16 * I + 8 * half) * NE + a1g * 16 + N;
        #pragma unroll
        for (int r = 0; r < 8; ++r) dst[r * NE] = (_Float16)d[r];
    }
    __syncthreads();

    // ---- stepB: smooth a1.  36 tiles: (I = a1-tile, a2g) ----
    _Float16* tb = t + (size_t)vol * NV + a0;
    #pragma unroll
    for (int it = 0; it < 4; ++it) {
        const int tt = wave + 9 * it;
        const int I = tt / 6, a2g = tt % 6;
        const int w0 = win0(I);
        const v16h afrag = load_afrag(Abc + I * ATILE_HALFS, lane);
        const _Float16* p = P1 + (a2g * 16 + N) * NE + w0 + 16 * half;
        const v16h bfrag = combine16(*(const v8h*)p, *(const v8h*)(p + 8));
        v8f cz = {};
        v8f d = __builtin_amdgcn_wmma_f32_16x16x32_f16(
                    false, afrag, false, bfrag, (short)0, cz, false, false);
        // t[a1][a2][a0]; a1 = 16I + 8*half + r, a2 = a2g*16 + N
        _Float16* dst = tb + (size_t)(16 * I + 8 * half) * S_YZ
                           + (a2g * 16 + N) * NE;
        #pragma unroll
        for (int r = 0; r < 8; ++r) dst[r * S_YZ] = (_Float16)d[r];
    }
}

// ---------------------------------------------------------------------------
// Pass 3: smooth a0 + fused einsum over K, fp32 out in natural layout.
// a0 is contiguous in t: line = a1*96 + a2, elem addr = line*96 + a0.
// out[b,c][a0][a1][a2] = sum_k D_k * x[b,k][a0][a1][a2]
// waves = 8 bc * 6 I * 72 = 3456 -> 432 blocks
// ---------------------------------------------------------------------------
__global__ __launch_bounds__(256) void pass3_kernel(
        const _Float16* __restrict__ t,
        const float* __restrict__ xresp,
        const _Float16* __restrict__ wsA,
        float* __restrict__ out) {
    const int lane = threadIdx.x & 31;
    const int w    = blockIdx.x * 8 + (threadIdx.x >> 5);
    const int blk  = w % 72;
    const int wI   = w / 72;
    const int I    = wI % NTILE;
    const int bc   = wI / NTILE;
    const int b    = bc / NC;

    const int w0   = win0(I);
    const int N    = lane & 15, half = lane >> 4;

    const v16h afrag = load_afrag(wsA + (bc * NTILE + I) * ATILE_HALFS, lane);

    for (int j = 0; j < 8; ++j) {
        const int line = (blk * 8 + j) * 16 + N;   // line = a1*96 + a2
        const int m0   = 16 * I + 8 * half;        // first a0 row this lane owns
        float acc[8];
        #pragma unroll
        for (int r = 0; r < 8; ++r) acc[r] = 0.0f;
        #pragma unroll
        for (int k = 0; k < NK; ++k) {
            const _Float16* p = t + (size_t)(bc * NK + k) * NV
                                  + line * NE + w0 + 16 * half;
            if (k + 1 < NK)
                __builtin_prefetch(p + NV, 0, 1);  // next K volume's fragment
            const v16h bfrag = combine16(*(const v8h*)p, *(const v8h*)(p + 8));
            v8f cz = {};
            v8f d = __builtin_amdgcn_wmma_f32_16x16x32_f16(
                        false, afrag, false, bfrag, (short)0, cz, false, false);
            const float* xp = xresp + (size_t)(b * NK + k) * NV + line;
            #pragma unroll
            for (int r = 0; r < 8; ++r)
                acc[r] = fmaf((float)d[r], xp[(m0 + r) * S_YZ], acc[r]);
        }
        float* op = out + (size_t)bc * NV + line;
        #pragma unroll
        for (int r = 0; r < 8; ++r)
            op[(m0 + r) * S_YZ] = acc[r];
    }
}

// ---------------------------------------------------------------------------
extern "C" void kernel_launch(void* const* d_in, const int* in_sizes, int n_in,
                              void* d_out, int out_size, void* d_ws, size_t ws_size,
                              hipStream_t stream) {
    (void)in_sizes; (void)n_in; (void)out_size; (void)ws_size;
    const float* x             = (const float*)d_in[0];  // (B,K,96^3)
    const float* means_eps     = (const float*)d_in[1];  // (B,C,K)
    const float* scales_sample = (const float*)d_in[2];  // (B,C,K)
    const float* fwhm_eps      = (const float*)d_in[3];  // (B,C)
    const float* noise         = (const float*)d_in[4];  // (B,C,K,96^3)
    float* out = (float*)d_out;                          // (B,C,96^3)

    _Float16* wsA = (_Float16*)d_ws;                     // 48KB S tiles
    _Float16* t   = (_Float16*)((char*)d_ws + 65536);    // 70.8MB f16

    build_S_kernel<<<NB * NC, 512, 0, stream>>>(fwhm_eps, wsA);
    fused12_kernel<<<NB * NC * NK * NE, 288, 0, stream>>>(
        noise, means_eps, scales_sample, wsA, t);
    pass3_kernel<<<432, 256, 0, stream>>>(t, x, wsA, out);
}
